// Adaptive_MultiGraph_Module_48601849922158
// MI455X (gfx1250) — matrix operand order repaired and checked by reading, not compiled, not run on hardware
//
#include <hip/hip_runtime.h>
#include <hip/hip_bf16.h>

typedef __attribute__((ext_vector_type(16))) __bf16 v16bf;
typedef __attribute__((ext_vector_type(8)))  float  v8f;
typedef __attribute__((ext_vector_type(4)))  float  f4v;
typedef __attribute__((ext_vector_type(4)))  unsigned int u32x4;
typedef __attribute__((ext_vector_type(4)))  int    i32x4;
typedef __attribute__((ext_vector_type(8)))  int    i32x8;

#define NN 4096
#define KCH 128   // K rows of B staged in LDS per TDM transfer

// TDM builtin arity differs between toolchains (5 args ROCm7.2 clang-22,
// 6 args amdgpu-toolchain clang-23 whose headers ship amd_gfx1250_TDM.h).
static __device__ __forceinline__ void tdm_load_to_lds(u32x4 g0, i32x8 g1) {
    i32x4 z4 = {0, 0, 0, 0};
#if __has_include(<hip/amd_detail/amd_gfx1250_TDM.h>)
    i32x8 z8 = {0, 0, 0, 0, 0, 0, 0, 0};
    __builtin_amdgcn_tensor_load_to_lds(g0, g1, z4, z4, z8, 0);
#else
    __builtin_amdgcn_tensor_load_to_lds(g0, g1, z4, z4, 0);
#endif
}

// ---------------------------------------------------------------------------
// dis[i] = rsqrt( 1 + sum_j binz(A[i][j]) )   (row-major adjacency)
// ---------------------------------------------------------------------------
__global__ void degree_rows_kernel(const float* __restrict__ A,
                                   float* __restrict__ dis) {
    __shared__ float red[256];
    const int row = blockIdx.x;
    float s = 0.f;
    const f4v* Ar = (const f4v*)(A + (size_t)row * NN);
    for (int j = threadIdx.x; j < NN / 4; j += 256) {
        f4v a = Ar[j];
        s += ((a[0] != 0.f) ? 1.f : 0.f) + ((a[1] != 0.f) ? 1.f : 0.f) +
             ((a[2] != 0.f) ? 1.f : 0.f) + ((a[3] != 0.f) ? 1.f : 0.f);
    }
    red[threadIdx.x] = s;
    __syncthreads();
    for (int off = 128; off > 0; off >>= 1) {
        if (threadIdx.x < off) red[threadIdx.x] += red[threadIdx.x + off];
        __syncthreads();
    }
    if (threadIdx.x == 0) dis[row] = rsqrtf(red[0] + 1.0f);   // +1 = identity
}

// ada graph: ada[i][j] = Wa[j][i] + ba[j]; thread-per-i -> coalesced Wa reads
__global__ void degree_ada_kernel(const float* __restrict__ Wa,
                                  const float* __restrict__ ba,
                                  float* __restrict__ dis) {
    const int i = blockIdx.x * blockDim.x + threadIdx.x;
    float s = 1.0f;                                           // identity
    for (int j = 0; j < NN; ++j) {
        float v = Wa[(size_t)j * NN + i] + ba[j];
        s += (v != 0.f) ? 1.f : 0.f;
    }
    dis[i] = rsqrtf(s);
}

// ---------------------------------------------------------------------------
// Out = [relu]( diag(dis) * (binz(A)+I) * diag(dis) @ B + bias )
// Row-major adjacency (graphs 0..3). One wave per 16-row strip, NT register
// tiles (Ncols = NT*16). A fragments from b128 loads; B staged in LDS via the
// Tensor Data Mover (one 2-D tile descriptor per KCH-row chunk).
// ---------------------------------------------------------------------------
template <int NT>
__global__ void norm_prop_rowmajor(const float* __restrict__ A,
                                   const float* __restrict__ dis,
                                   const float* __restrict__ B,   // NN x NT*16
                                   const float* __restrict__ bias,
                                   float* __restrict__ Out,
                                   int doRelu) {
    constexpr int Ncols = NT * 16;
    __shared__ float smemB[KCH * Ncols];      // first (only) LDS alloc -> offset 0

    const int tid  = threadIdx.x;
    const int wave = tid >> 5;
    const int lane = tid & 31;
    const int hf   = lane >> 4;          // which half of the wave
    const int lrow = lane & 15;
    const int mBase = (blockIdx.x * 4 + wave) * 16;
    const int row   = mBase + lrow;

    const float* Arow = A + (size_t)row * NN;

    v8f acc[NT];
#pragma unroll
    for (int nt = 0; nt < NT; ++nt)
        acc[nt] = (v8f){0.f, 0.f, 0.f, 0.f, 0.f, 0.f, 0.f, 0.f};

    for (int kc = 0; kc < NN; kc += KCH) {
        // ---- stage B[kc .. kc+KCH) x Ncols into LDS with the TDM ---------
        __syncthreads();                      // previous chunk fully consumed
        if (wave == 0) {
            const unsigned long long gaddr =
                (unsigned long long)(const void*)(B + (size_t)kc * Ncols);
            u32x4 g0;
            g0[0] = 1u;                                   // count=1 (valid D#)
            g0[1] = 0u;                                   // lds_addr = 0 (smemB)
            g0[2] = (unsigned)(gaddr & 0xFFFFFFFFu);      // global_addr[31:0]
            g0[3] = (unsigned)((gaddr >> 32) & 0x01FFFFFFu) | (2u << 30); // type=2
            i32x8 g1;
            g1[0] = (int)(2u << 16);                      // data_size = 4B
            g1[1] = (int)((unsigned)Ncols << 16);         // tensor_dim0[15:0]
            g1[2] = (int)(((unsigned)NN & 0xFFFFu) << 16);// dim0 hi=0 | tensor_dim1 lo
            g1[3] = (int)(((unsigned)NN >> 16) | ((unsigned)Ncols << 16)); // dim1 hi | tile_dim0
            g1[4] = (int)(KCH & 0xFFFF);                  // tile_dim1 = KCH rows
            g1[5] = (int)Ncols;                           // tensor_dim0_stride lo32
            g1[6] = 0;
            g1[7] = 0;
            tdm_load_to_lds(g0, g1);
            __builtin_amdgcn_s_wait_tensorcnt(0);
        }
        __syncthreads();                      // LDS tile visible to all waves

        for (int k0 = kc; k0 < kc + KCH; k0 += 32) {
            if (k0 + 32 < NN) __builtin_prefetch(Arow + k0 + 32, 0, 0);

            // ---- A fragment: 16x32 bf16 (ISA layout), b128 loads ---------
            v16bf afrag;
#pragma unroll
            for (int g = 0; g < 2; ++g) {
                const int kb = k0 + g * 16 + hf * 8;
                const f4v a0 = *(const f4v*)(Arow + kb);
                const f4v a1 = *(const f4v*)(Arow + kb + 4);
                const f4v d0 = *(const f4v*)(dis + kb);
                const f4v d1 = *(const f4v*)(dis + kb + 4);
#pragma unroll
                for (int i = 0; i < 4; ++i) {
                    float av = ((a0[i] != 0.f) ? 1.f : 0.f) +
                               ((kb + i == row) ? 1.f : 0.f);
                    afrag[g * 8 + i] = (__bf16)(av * d0[i]);
                }
#pragma unroll
                for (int i = 0; i < 4; ++i) {
                    float av = ((a1[i] != 0.f) ? 1.f : 0.f) +
                               ((kb + 4 + i == row) ? 1.f : 0.f);
                    afrag[g * 8 + 4 + i] = (__bf16)(av * d1[i]);
                }
            }

            // ---- NT B fragments (from LDS) + NT WMMAs --------------------
            const int klocal = (k0 - kc) + (hf << 3);
#pragma unroll
            for (int nt = 0; nt < NT; ++nt) {
                v16bf bfrag;
#pragma unroll
                for (int e = 0; e < 16; ++e)
                    bfrag[e] = (__bf16)smemB[(klocal + (e & 7) + ((e >> 3) << 4)) * Ncols + nt * 16 + lrow];
                acc[nt] = __builtin_amdgcn_wmma_f32_16x16x32_bf16(
                    false, afrag, false, bfrag, (short)0, acc[nt], false, false);
            }
        }
    }

    // ---- Epilogue: row normalization, bias, optional relu ----------------
    float dism[8];
#pragma unroll
    for (int r = 0; r < 8; ++r) dism[r] = dis[mBase + r + hf * 8];
#pragma unroll
    for (int nt = 0; nt < NT; ++nt) {
        const int colN = nt * 16 + lrow;
        const float bval = bias[colN];
#pragma unroll
        for (int r = 0; r < 8; ++r) {
            const int m = mBase + r + hf * 8;
            float v = dism[r] * acc[nt][r] + bval;
            if (doRelu) v = fmaxf(v, 0.f);
            Out[(size_t)m * Ncols + colN] = v;
        }
    }
}

// ---------------------------------------------------------------------------
// Same, for the ada graph: Ah(r,k) = binz(Wa[k][r] + ba[k]) + (r==k)
// (column-strided Wa reads; straight-line, branchless; B from global)
// ---------------------------------------------------------------------------
template <int NT>
__global__ void norm_prop_ada(const float* __restrict__ Wa,
                              const float* __restrict__ ba,
                              const float* __restrict__ dis,
                              const float* __restrict__ B,
                              const float* __restrict__ bias,
                              float* __restrict__ Out,
                              int doRelu) {
    constexpr int Ncols = NT * 16;
    const int tid  = threadIdx.x;
    const int wave = tid >> 5;
    const int lane = tid & 31;
    const int hf   = lane >> 4;
    const int lrow = lane & 15;
    const int mBase = (blockIdx.x * 4 + wave) * 16;
    const int row   = mBase + lrow;

    v8f acc[NT];
#pragma unroll
    for (int nt = 0; nt < NT; ++nt)
        acc[nt] = (v8f){0.f, 0.f, 0.f, 0.f, 0.f, 0.f, 0.f, 0.f};

    for (int k0 = 0; k0 < NN; k0 += 32) {
        v16bf afrag;
#pragma unroll
        for (int g = 0; g < 2; ++g) {
            const int kb = k0 + g * 16 + hf * 8;
            const f4v d0 = *(const f4v*)(dis + kb);
            const f4v d1 = *(const f4v*)(dis + kb + 4);
            const f4v c0 = *(const f4v*)(ba + kb);
            const f4v c1 = *(const f4v*)(ba + kb + 4);
#pragma unroll
            for (int i = 0; i < 8; ++i) {
                const int kk = kb + i;
                const float w = Wa[(size_t)kk * NN + row] +
                                (i < 4 ? c0[i & 3] : c1[i & 3]);
                float av = ((w != 0.f) ? 1.f : 0.f) + ((kk == row) ? 1.f : 0.f);
                const float d = (i < 4) ? d0[i & 3] : d1[i & 3];
                afrag[g * 8 + i] = (__bf16)(av * d);
            }
        }
#pragma unroll
        for (int nt = 0; nt < NT; ++nt) {
            v16bf bfrag;
#pragma unroll
            for (int e = 0; e < 16; ++e) {
                const int kk = k0 + (hf << 3) + (e & 7) + ((e >> 3) << 4);
                bfrag[e] = (__bf16)B[(size_t)kk * Ncols + nt * 16 + lrow];
            }
            acc[nt] = __builtin_amdgcn_wmma_f32_16x16x32_bf16(
                false, afrag, false, bfrag, (short)0, acc[nt], false, false);
        }
    }

    float dism[8];
#pragma unroll
    for (int r = 0; r < 8; ++r) dism[r] = dis[mBase + r + hf * 8];
#pragma unroll
    for (int nt = 0; nt < NT; ++nt) {
        const int colN = nt * 16 + lrow;
        const float bval = bias[colN];
#pragma unroll
        for (int r = 0; r < 8; ++r) {
            const int m = mBase + r + hf * 8;
            float v = dism[r] * acc[nt][r] + bval;
            if (doRelu) v = fmaxf(v, 0.f);
            Out[(size_t)m * Ncols + colN] = v;
        }
    }
}

// ---------------------------------------------------------------------------
// Dense C(M x NT*16) = A(M x K, row-major, lda) @ B(K x NT*16)
// ---------------------------------------------------------------------------
template <int NT>
__global__ void dense_gemm_wmma(const float* __restrict__ A, int lda,
                                const float* __restrict__ B,
                                float* __restrict__ C, int K) {
    constexpr int Ncols = NT * 16;
    const int tid  = threadIdx.x;
    const int wave = tid >> 5;
    const int lane = tid & 31;
    const int hf   = lane >> 4;
    const int lrow = lane & 15;
    const int mBase = (blockIdx.x * 4 + wave) * 16;
    const float* Arow = A + (size_t)(mBase + lrow) * lda;

    v8f acc[NT];
#pragma unroll
    for (int nt = 0; nt < NT; ++nt)
        acc[nt] = (v8f){0.f, 0.f, 0.f, 0.f, 0.f, 0.f, 0.f, 0.f};

    for (int k0 = 0; k0 < K; k0 += 32) {
        v16bf afrag;
#pragma unroll
        for (int g = 0; g < 2; ++g) {
            const int kb = k0 + g * 16 + hf * 8;
            const f4v a0 = *(const f4v*)(Arow + kb);
            const f4v a1 = *(const f4v*)(Arow + kb + 4);
#pragma unroll
            for (int i = 0; i < 4; ++i) afrag[g * 8 + i]     = (__bf16)a0[i];
#pragma unroll
            for (int i = 0; i < 4; ++i) afrag[g * 8 + 4 + i] = (__bf16)a1[i];
        }
#pragma unroll
        for (int nt = 0; nt < NT; ++nt) {
            v16bf bfrag;
#pragma unroll
            for (int e = 0; e < 16; ++e) {
                const int kk = k0 + (hf << 3) + (e & 7) + ((e >> 3) << 4);
                bfrag[e] = (__bf16)B[(size_t)kk * Ncols + nt * 16 + lrow];
            }
            acc[nt] = __builtin_amdgcn_wmma_f32_16x16x32_bf16(
                false, afrag, false, bfrag, (short)0, acc[nt], false, false);
        }
    }
#pragma unroll
    for (int nt = 0; nt < NT; ++nt)
#pragma unroll
        for (int r = 0; r < 8; ++r)
            C[(size_t)(mBase + r + hf * 8) * Ncols + nt * 16 + lrow] = acc[nt][r];
}

// ---------------------------------------------------------------------------
// out[r] = dot(W[r, :K], x) + b[r]   (one block per output row, float4 loads)
// ---------------------------------------------------------------------------
__global__ void matvec_kernel(const float* __restrict__ W,
                              const float* __restrict__ x,
                              const float* __restrict__ b,
                              float* __restrict__ out, int K) {
    __shared__ float red[256];
    const int r = blockIdx.x;
    const f4v* W4 = (const f4v*)(W + (size_t)r * K);
    const f4v* x4 = (const f4v*)x;
    float s = 0.f;
    for (int j = threadIdx.x; j < K / 4; j += 256) {
        const f4v w = W4[j], v = x4[j];
        s += w[0] * v[0] + w[1] * v[1] + w[2] * v[2] + w[3] * v[3];
    }
    red[threadIdx.x] = s;
    __syncthreads();
    for (int off = 128; off > 0; off >>= 1) {
        if (threadIdx.x < off) red[threadIdx.x] += red[threadIdx.x + off];
        __syncthreads();
    }
    if (threadIdx.x == 0) out[r] = red[0] + b[r];
}

// ---------------------------------------------------------------------------
// Fusion + attention tail (tiny): one block, 64 threads.
// ---------------------------------------------------------------------------
__global__ void tail_kernel(const float* __restrict__ feats,   // 5 x 32
                            const float* __restrict__ WvF, const float* __restrict__ bvF,
                            const float* __restrict__ WoF, const float* __restrict__ boF,
                            const float* __restrict__ WvA, const float* __restrict__ bvA,
                            const float* __restrict__ WoA, const float* __restrict__ boA,
                            float* __restrict__ out) {
    __shared__ float v[32], cat[64], V[64];
    const int t = threadIdx.x;
    if (t < 32) {                       // v = feats[2] @ WvF.T + bvF
        float s = bvF[t];
        for (int j = 0; j < 32; ++j) s += feats[2 * 32 + j] * WvF[t * 32 + j];
        v[t] = s;
    }
    __syncthreads();
    if (t < 32) {                       // fusion = v @ WoF.T + boF
        float s = boF[t];
        for (int j = 0; j < 32; ++j) s += v[j] * WoF[t * 32 + j];
        cat[t] = s;
    } else {
        cat[t] = feats[4 * 32 + (t - 32)];   // out_ada
    }
    __syncthreads();
    {                                   // V = cat @ WvA.T + bvA   (64x64)
        float s = bvA[t];
        for (int j = 0; j < 64; ++j) s += cat[j] * WvA[t * 64 + j];
        V[t] = s;
    }
    __syncthreads();
    if (t < 32) {                       // out = V @ WoA.T + boA   (32x64)
        float s = boA[t];
        for (int j = 0; j < 64; ++j) s += V[j] * WoA[t * 64 + j];
        out[t] = s;
    }
}

// ---------------------------------------------------------------------------
extern "C" void kernel_launch(void* const* d_in, const int* in_sizes, int n_in,
                              void* d_out, int out_size, void* d_ws, size_t ws_size,
                              hipStream_t stream) {
    (void)in_sizes; (void)n_in; (void)out_size; (void)ws_size;

    // workspace layout (floats)
    float* ws    = (float*)d_ws;
    float* dis   = ws;                      // 4096
    float* H     = dis + NN;                // 4096*64
    float* T     = H + (size_t)NN * 64;     // 4096*32
    float* H2    = T + (size_t)NN * 32;     // 4096*32
    float* feats = H2 + (size_t)NN * 32;    // 5*32

    const float* Wa = (const float*)d_in[34];
    const float* ba = (const float*)d_in[35];

    for (int g = 0; g < 5; ++g) {
        const int pb = (g < 4) ? (4 + g * 6) : 28;
        const float* W1 = (const float*)d_in[pb + 0];
        const float* b1 = (const float*)d_in[pb + 1];
        const float* W2 = (const float*)d_in[pb + 2];
        const float* b2 = (const float*)d_in[pb + 3];
        const float* Wl = (const float*)d_in[pb + 4];
        const float* bl = (const float*)d_in[pb + 5];

        if (g < 4) {
            const float* A = (const float*)d_in[g];
            degree_rows_kernel<<<NN, 256, 0, stream>>>(A, dis);
            // H = relu(norm @ W1 + b1)        (4096 x 64)
            norm_prop_rowmajor<4><<<NN / 64, 128, 0, stream>>>(A, dis, W1, b1, H, 1);
            // T = H @ W2                      (4096 x 32)
            dense_gemm_wmma<2><<<NN / 64, 128, 0, stream>>>(H, 64, W2, T, 64);
            // H2 = norm @ T + b2              (4096 x 32)
            norm_prop_rowmajor<2><<<NN / 64, 128, 0, stream>>>(A, dis, T, b2, H2, 0);
        } else {
            degree_ada_kernel<<<NN / 256, 256, 0, stream>>>(Wa, ba, dis);
            norm_prop_ada<4><<<NN / 64, 128, 0, stream>>>(Wa, ba, dis, W1, b1, H, 1);
            dense_gemm_wmma<2><<<NN / 64, 128, 0, stream>>>(H, 64, W2, T, 64);
            norm_prop_ada<2><<<NN / 64, 128, 0, stream>>>(Wa, ba, dis, T, b2, H2, 0);
        }
        // feats[g] = Wl @ vec(H2) + bl        (32)
        matvec_kernel<<<32, 256, 0, stream>>>(Wl, H2, bl, feats + g * 32, NN * 32);
    }

    // p_fus: Wv=d_in[40], bv=41, Wo=42, bo=43 ; p_att: Wv=48, bv=49, Wo=50, bo=51
    tail_kernel<<<1, 64, 0, stream>>>(feats,
        (const float*)d_in[40], (const float*)d_in[41],
        (const float*)d_in[42], (const float*)d_in[43],
        (const float*)d_in[48], (const float*)d_in[49],
        (const float*)d_in[50], (const float*)d_in[51],
        (float*)d_out);
}
